// GraphEmbed_69011534512457
// MI455X (gfx1250) — matrix operand order repaired and checked by reading, not compiled, not run on hardware
//
#include <hip/hip_runtime.h>
#include <cstdint>
#include <cstddef>

// Problem constants (match reference)
#define B_      4
#define N_      4096
#define E_      65536
#define HID_    128
#define FN_     32
#define FE_     16
#define ROUNDS_ 3

// CDNA5 WMMA vector types (wave32)
typedef __attribute__((ext_vector_type(16))) __bf16 v16bf;
typedef __attribute__((ext_vector_type(8)))  float  v8f;

union ABfrag { v16bf v; unsigned int w[8]; unsigned short h[16]; };

__device__ __forceinline__ unsigned short f2bf(float f) {
  unsigned int x = __float_as_uint(f);
  unsigned int r = x + 0x7fffu + ((x >> 16) & 1u);   // RNE
  return (unsigned short)(r >> 16);
}
__device__ __forceinline__ float siluf(float x) {
  return x / (1.f + __expf(-x));
}

// Padded K strides (elements) for LDS-resident transposed weights.
#define KP1E 392   // for K=384 (16B-aligned rows, bank-spread)
#define KP1N 264   // for K=256
#define KP2  136   // for K=128 (also hidden staging)

// ---------------------------------------------------------------------------
// One-time: W (K,128) f32 row-major  ->  W^T bf16 [n][k] with padded stride.
// ---------------------------------------------------------------------------
__global__ void transpose_w_kernel(const float* __restrict__ src,
                                   unsigned short* __restrict__ dst,
                                   int K, int KP) {
  int total = K * HID_;
  for (int i = blockIdx.x * blockDim.x + threadIdx.x; i < total;
       i += gridDim.x * blockDim.x) {
    int k = i >> 7, n = i & 127;
    dst[n * KP + k] = f2bf(src[i]);
  }
}

// ---------------------------------------------------------------------------
// h = nodes @ W_node  (K=32, tiny) -> h_f32 + h_bf16
// ---------------------------------------------------------------------------
__global__ void embed_nodes_kernel(const float* __restrict__ nodes,
                                   const float* __restrict__ Wn,
                                   float* __restrict__ h_f32,
                                   unsigned short* __restrict__ h_bf) {
  const long long total = (long long)B_ * N_ * HID_;
  for (long long i = (long long)blockIdx.x * blockDim.x + threadIdx.x;
       i < total; i += (long long)gridDim.x * blockDim.x) {
    long long row = i >> 7; int c = (int)(i & 127);
    const float* nr = nodes + row * FN_;
    float acc = 0.f;
#pragma unroll
    for (int j = 0; j < FN_; ++j) acc += nr[j] * Wn[j * HID_ + c];
    h_f32[i] = acc;
    h_bf[i]  = f2bf(acc);
  }
}

// ---------------------------------------------------------------------------
// ef = edge_feat @ W_edge  (K=16) -> bf16 (reused across all 3 rounds)
// ---------------------------------------------------------------------------
__global__ void embed_edges_kernel(const float* __restrict__ ef_in,
                                   const float* __restrict__ We,
                                   unsigned short* __restrict__ ef_bf) {
  const long long total = (long long)B_ * E_ * HID_;
  for (long long i = (long long)blockIdx.x * blockDim.x + threadIdx.x;
       i < total; i += (long long)gridDim.x * blockDim.x) {
    long long row = i >> 7; int c = (int)(i & 127);
    const float* er = ef_in + row * FE_;
    float acc = 0.f;
#pragma unroll
    for (int j = 0; j < FE_; ++j) acc += er[j] * We[j * HID_ + c];
    ef_bf[i] = f2bf(acc);
  }
}

__global__ void zero_kernel(float* __restrict__ p, long long n) {
  for (long long i = (long long)blockIdx.x * blockDim.x + threadIdx.x;
       i < n; i += (long long)gridDim.x * blockDim.x) p[i] = 0.f;
}

// ---------------------------------------------------------------------------
// Fused edge MLP + scatter-add.  One wave per 16-edge tile.
// A fragments gathered straight from global bf16; pre-transposed bf16 weights
// copied contiguously into LDS; per-K-block all 8 B fragments are preloaded
// into registers so the 8 WMMAs issue back-to-back behind a single DS wait.
// ---------------------------------------------------------------------------
__global__ __launch_bounds__(256) void edge_mlp_kernel(
    const unsigned short* __restrict__ h_bf,
    const unsigned short* __restrict__ ef_bf,
    const long long* __restrict__ edges,
    const unsigned char* __restrict__ emask,
    const unsigned short* __restrict__ W1T,   // [128][KP1E] bf16
    const float* __restrict__ b1,
    const unsigned short* __restrict__ W2T,   // [128][KP2] bf16
    const float* __restrict__ b2,
    float* __restrict__ agg) {
  __shared__ __align__(16) unsigned short sW1[HID_ * KP1E];
  __shared__ __align__(16) unsigned short sW2[HID_ * KP2];
  __shared__ float sB1[HID_], sB2[HID_];
  __shared__ __align__(16) unsigned short sHid[8 * 16 * KP2];

  const int tid = threadIdx.x;
  {
    const uint4* s4 = (const uint4*)W1T;  uint4* d4 = (uint4*)sW1;
    for (int i = tid; i < (HID_ * KP1E) / 8; i += 256) d4[i] = s4[i];
    s4 = (const uint4*)W2T;  d4 = (uint4*)sW2;
    for (int i = tid; i < (HID_ * KP2) / 8; i += 256) d4[i] = s4[i];
  }
  if (tid < HID_) { sB1[tid] = b1[tid]; sB2[tid] = b2[tid]; }
  __syncthreads();

  const int lane = tid & 31;
  const int wave = tid >> 5;
  const int half = lane >> 4;   // K-half selector (A/B layouts)
  const int mr   = lane & 15;   // A row owned by this lane
  const int ncol = lane & 15;   // C/D column owned by this lane
  unsigned short* hid = &sHid[wave * 16 * KP2];

  const int nTiles   = (B_ * E_) / 16;
  const int waveGlob = blockIdx.x * 8 + wave;
  const int waveCnt  = gridDim.x * 8;

  for (int tile = waveGlob; tile < nTiles; tile += waveCnt) {
    const long long fe = (long long)tile * 16 + mr;
    const int b = (int)(fe >> 16);                     // E_ = 65536
    long long sv = edges[fe * 2 + 0];
    long long dv = edges[fe * 2 + 1];
    int s = (int)(sv < 0 ? 0 : (sv > N_ - 1 ? N_ - 1 : sv));
    int d = (int)(dv < 0 ? 0 : (dv > N_ - 1 ? N_ - 1 : dv));
    const unsigned short* rowS = h_bf + ((size_t)b * N_ + s) * HID_;
    const unsigned short* rowD = h_bf + ((size_t)b * N_ + d) * HID_;
    const unsigned short* rowE = ef_bf + (size_t)fe * HID_;

    // ---- layer 1: hidden = silu([h_src|h_dst|ef] @ W1 + b1) ----
    v8f acc[8];
#pragma unroll
    for (int nt = 0; nt < 8; ++nt) {
      float bv = sB1[nt * 16 + ncol];
      v8f t = {bv, bv, bv, bv, bv, bv, bv, bv};
      acc[nt] = t;
    }
#pragma unroll
    for (int kb = 0; kb < 12; ++kb) {
      const unsigned short* row = (kb < 4) ? rowS : (kb < 8 ? rowD : rowE);
      const int kofs = (kb & 3) * 32;
      ABfrag a;
#pragma unroll
      for (int v = 0; v < 8; ++v) {
        int k0 = kofs + 2 * v + 16 * half;
        a.w[v] = *(const unsigned int*)(row + k0);
      }
      ABfrag bf[8];
#pragma unroll
      for (int nt = 0; nt < 8; ++nt) {
        const unsigned short* wr = &sW1[(nt * 16 + ncol) * KP1E + kb * 32 + 16 * half];
#pragma unroll
        for (int v = 0; v < 8; ++v) bf[nt].w[v] = *(const unsigned int*)(wr + 2 * v);
      }
#pragma unroll
      for (int nt = 0; nt < 8; ++nt)
        acc[nt] = __builtin_amdgcn_wmma_f32_16x16x32_bf16(
            false, a.v, false, bf[nt].v, (short)0, acc[nt], false, false);
    }
    // SiLU + C-layout -> A-layout transpose via per-wave LDS
#pragma unroll
    for (int nt = 0; nt < 8; ++nt) {
#pragma unroll
      for (int g = 0; g < 8; ++g) {
        int m = g + 8 * half;
        hid[m * KP2 + nt * 16 + ncol] = f2bf(siluf(acc[nt][g]));
      }
    }
    // ---- layer 2: msg = hidden @ W2 + b2 ----
#pragma unroll
    for (int nt = 0; nt < 8; ++nt) {
      float bv = sB2[nt * 16 + ncol];
      v8f t = {bv, bv, bv, bv, bv, bv, bv, bv};
      acc[nt] = t;
    }
#pragma unroll
    for (int kb = 0; kb < 4; ++kb) {
      ABfrag a;
      const unsigned short* row = hid + mr * KP2 + kb * 32;
#pragma unroll
      for (int v = 0; v < 8; ++v) {
        int k0 = 2 * v + 16 * half;
        a.w[v] = *(const unsigned int*)(row + k0);
      }
      ABfrag bf[8];
#pragma unroll
      for (int nt = 0; nt < 8; ++nt) {
        const unsigned short* wr = &sW2[(nt * 16 + ncol) * KP2 + kb * 32 + 16 * half];
#pragma unroll
        for (int v = 0; v < 8; ++v) bf[nt].w[v] = *(const unsigned int*)(wr + 2 * v);
      }
#pragma unroll
      for (int nt = 0; nt < 8; ++nt)
        acc[nt] = __builtin_amdgcn_wmma_f32_16x16x32_bf16(
            false, a.v, false, bf[nt].v, (short)0, acc[nt], false, false);
    }
    // ---- mask + scatter-add (hardware f32 atomics) ----
    int bRow[8], dRow[8]; float mk[8];
#pragma unroll
    for (int g = 0; g < 8; ++g) {
      long long feg = (long long)tile * 16 + g + 8 * half;
      bRow[g] = (int)(feg >> 16);
      long long dvg = edges[feg * 2 + 1];
      dRow[g] = (int)(dvg < 0 ? 0 : (dvg > N_ - 1 ? N_ - 1 : dvg));
      mk[g] = emask[feg] ? 1.f : 0.f;
    }
#pragma unroll
    for (int nt = 0; nt < 8; ++nt) {
#pragma unroll
      for (int g = 0; g < 8; ++g) {
        float val = acc[nt][g] * mk[g];
        float* p = agg + ((size_t)bRow[g] * N_ + dRow[g]) * HID_ + nt * 16 + ncol;
        unsafeAtomicAdd(p, val);
      }
    }
  }
}

// ---------------------------------------------------------------------------
// Fused node MLP: h = (h + silu([h|agg]@nW1+nb1)@nW2+nb2) * node_mask
// In-place h update is safe: each 16-row tile touches only its own rows.
// ---------------------------------------------------------------------------
__global__ __launch_bounds__(256) void node_mlp_kernel(
    float* __restrict__ h_f32,
    unsigned short* __restrict__ h_bf,
    const float* __restrict__ agg,
    const unsigned char* __restrict__ nmask,
    const unsigned short* __restrict__ W1T,   // [128][KP1N] bf16
    const float* __restrict__ b1,
    const unsigned short* __restrict__ W2T,   // [128][KP2] bf16
    const float* __restrict__ b2) {
  __shared__ __align__(16) unsigned short sW1[HID_ * KP1N];
  __shared__ __align__(16) unsigned short sW2[HID_ * KP2];
  __shared__ float sB1[HID_], sB2[HID_];
  __shared__ __align__(16) unsigned short sHid[8 * 16 * KP2];

  const int tid = threadIdx.x;
  {
    const uint4* s4 = (const uint4*)W1T;  uint4* d4 = (uint4*)sW1;
    for (int i = tid; i < (HID_ * KP1N) / 8; i += 256) d4[i] = s4[i];
    s4 = (const uint4*)W2T;  d4 = (uint4*)sW2;
    for (int i = tid; i < (HID_ * KP2) / 8; i += 256) d4[i] = s4[i];
  }
  if (tid < HID_) { sB1[tid] = b1[tid]; sB2[tid] = b2[tid]; }
  __syncthreads();

  const int lane = tid & 31;
  const int wave = tid >> 5;
  const int half = lane >> 4;
  const int mr   = lane & 15;
  const int ncol = lane & 15;
  unsigned short* hid = &sHid[wave * 16 * KP2];

  const int nTiles   = (B_ * N_) / 16;
  const int waveGlob = blockIdx.x * 8 + wave;
  const int waveCnt  = gridDim.x * 8;

  for (int tile = waveGlob; tile < nTiles; tile += waveCnt) {
    const size_t r0 = (size_t)tile * 16 + mr;           // flat row b*N+n
    const unsigned short* rowH = h_bf + r0 * HID_;
    const float*          rowA = agg  + r0 * HID_;

    v8f acc[8];
#pragma unroll
    for (int nt = 0; nt < 8; ++nt) {
      float bv = sB1[nt * 16 + ncol];
      v8f t = {bv, bv, bv, bv, bv, bv, bv, bv};
      acc[nt] = t;
    }
#pragma unroll
    for (int kb = 0; kb < 8; ++kb) {
      ABfrag a;
      if (kb < 4) {
        const int kofs = kb * 32;
#pragma unroll
        for (int v = 0; v < 8; ++v) {
          int k0 = kofs + 2 * v + 16 * half;
          a.w[v] = *(const unsigned int*)(rowH + k0);
        }
      } else {
        const int kofs = (kb - 4) * 32;
#pragma unroll
        for (int v = 0; v < 8; ++v) {
          int k0 = kofs + 2 * v + 16 * half;
          a.h[2 * v]     = f2bf(rowA[k0]);
          a.h[2 * v + 1] = f2bf(rowA[k0 + 1]);
        }
      }
      ABfrag bf[8];
#pragma unroll
      for (int nt = 0; nt < 8; ++nt) {
        const unsigned short* wr = &sW1[(nt * 16 + ncol) * KP1N + kb * 32 + 16 * half];
#pragma unroll
        for (int v = 0; v < 8; ++v) bf[nt].w[v] = *(const unsigned int*)(wr + 2 * v);
      }
#pragma unroll
      for (int nt = 0; nt < 8; ++nt)
        acc[nt] = __builtin_amdgcn_wmma_f32_16x16x32_bf16(
            false, a.v, false, bf[nt].v, (short)0, acc[nt], false, false);
    }
#pragma unroll
    for (int nt = 0; nt < 8; ++nt) {
#pragma unroll
      for (int g = 0; g < 8; ++g) {
        int m = g + 8 * half;
        hid[m * KP2 + nt * 16 + ncol] = f2bf(siluf(acc[nt][g]));
      }
    }
#pragma unroll
    for (int nt = 0; nt < 8; ++nt) {
      float bv = sB2[nt * 16 + ncol];
      v8f t = {bv, bv, bv, bv, bv, bv, bv, bv};
      acc[nt] = t;
    }
#pragma unroll
    for (int kb = 0; kb < 4; ++kb) {
      ABfrag a;
      const unsigned short* row = hid + mr * KP2 + kb * 32;
#pragma unroll
      for (int v = 0; v < 8; ++v) {
        int k0 = 2 * v + 16 * half;
        a.w[v] = *(const unsigned int*)(row + k0);
      }
      ABfrag bf[8];
#pragma unroll
      for (int nt = 0; nt < 8; ++nt) {
        const unsigned short* wr = &sW2[(nt * 16 + ncol) * KP2 + kb * 32 + 16 * half];
#pragma unroll
        for (int v = 0; v < 8; ++v) bf[nt].w[v] = *(const unsigned int*)(wr + 2 * v);
      }
#pragma unroll
      for (int nt = 0; nt < 8; ++nt)
        acc[nt] = __builtin_amdgcn_wmma_f32_16x16x32_bf16(
            false, a.v, false, bf[nt].v, (short)0, acc[nt], false, false);
    }
    // residual + node mask, write back f32 and bf16 copies
#pragma unroll
    for (int nt = 0; nt < 8; ++nt) {
#pragma unroll
      for (int g = 0; g < 8; ++g) {
        size_t row = (size_t)tile * 16 + g + 8 * half;
        size_t idx = row * HID_ + nt * 16 + ncol;
        float nm = nmask[row] ? 1.f : 0.f;
        float out = (h_f32[idx] + acc[nt][g]) * nm;
        h_f32[idx] = out;
        h_bf[idx]  = f2bf(out);
      }
    }
  }
}

// ---------------------------------------------------------------------------
// out[b,0,:] = marker ; out[b,1+n,:] = h[b,n,:]
// ---------------------------------------------------------------------------
__global__ void finalize_kernel(const float* __restrict__ h_f32,
                                const float* __restrict__ marker,
                                float* __restrict__ out) {
  const long long total = (long long)B_ * (N_ + 1) * HID_;
  for (long long i = (long long)blockIdx.x * blockDim.x + threadIdx.x;
       i < total; i += (long long)gridDim.x * blockDim.x) {
    int c = (int)(i % HID_);
    long long rowAll = i / HID_;
    long long b = rowAll / (N_ + 1);
    long long r = rowAll % (N_ + 1);
    out[i] = (r == 0) ? marker[c]
                      : h_f32[((size_t)b * N_ + (size_t)(r - 1)) * HID_ + c];
  }
}

// ---------------------------------------------------------------------------
extern "C" void kernel_launch(void* const* d_in, const int* in_sizes, int n_in,
                              void* d_out, int out_size, void* d_ws, size_t ws_size,
                              hipStream_t stream) {
  (void)in_sizes; (void)n_in; (void)out_size; (void)ws_size;
  const float*          nodes     = (const float*)d_in[0];
  const float*          edge_feat = (const float*)d_in[1];
  const long long*      edges     = (const long long*)d_in[2];
  const unsigned char*  node_mask = (const unsigned char*)d_in[3];
  const unsigned char*  edge_mask = (const unsigned char*)d_in[4];
  const float*          W_node    = (const float*)d_in[5];
  const float*          W_edge    = (const float*)d_in[6];
  const float*          marker    = (const float*)d_in[7];
  const float*          eW1       = (const float*)d_in[8];
  const float*          eb1       = (const float*)d_in[9];
  const float*          eW2       = (const float*)d_in[10];
  const float*          eb2       = (const float*)d_in[11];
  const float*          nW1       = (const float*)d_in[12];
  const float*          nb1       = (const float*)d_in[13];
  const float*          nW2       = (const float*)d_in[14];
  const float*          nb2       = (const float*)d_in[15];

  char* w = (char*)d_ws;
  float* h_f32 = (float*)w;                   w += (size_t)B_ * N_ * HID_ * sizeof(float);
  unsigned short* h_bf = (unsigned short*)w;  w += (size_t)B_ * N_ * HID_ * 2;
  unsigned short* ef_bf = (unsigned short*)w; w += (size_t)B_ * E_ * HID_ * 2;
  float* agg = (float*)w;                     w += (size_t)B_ * N_ * HID_ * sizeof(float);
  unsigned short* eW1T = (unsigned short*)w;  w += (size_t)ROUNDS_ * HID_ * KP1E * 2;
  unsigned short* eW2T = (unsigned short*)w;  w += (size_t)ROUNDS_ * HID_ * KP2 * 2;
  unsigned short* nW1T = (unsigned short*)w;  w += (size_t)ROUNDS_ * HID_ * KP1N * 2;
  unsigned short* nW2T = (unsigned short*)w;  w += (size_t)ROUNDS_ * HID_ * KP2 * 2;

  // one-time weight transposition to bf16 fragment-friendly layout
  for (int r = 0; r < ROUNDS_; ++r) {
    transpose_w_kernel<<<96, 256, 0, stream>>>(
        eW1 + (size_t)r * 3 * HID_ * HID_, eW1T + (size_t)r * HID_ * KP1E, 3 * HID_, KP1E);
    transpose_w_kernel<<<32, 256, 0, stream>>>(
        eW2 + (size_t)r * HID_ * HID_, eW2T + (size_t)r * HID_ * KP2, HID_, KP2);
    transpose_w_kernel<<<64, 256, 0, stream>>>(
        nW1 + (size_t)r * 2 * HID_ * HID_, nW1T + (size_t)r * HID_ * KP1N, 2 * HID_, KP1N);
    transpose_w_kernel<<<32, 256, 0, stream>>>(
        nW2 + (size_t)r * HID_ * HID_, nW2T + (size_t)r * HID_ * KP2, HID_, KP2);
  }

  embed_nodes_kernel<<<1024, 256, 0, stream>>>(nodes, W_node, h_f32, h_bf);
  embed_edges_kernel<<<4096, 256, 0, stream>>>(edge_feat, W_edge, ef_bf);

  for (int r = 0; r < ROUNDS_; ++r) {
    zero_kernel<<<1024, 256, 0, stream>>>(agg, (long long)B_ * N_ * HID_);
    edge_mlp_kernel<<<512, 256, 0, stream>>>(
        h_bf, ef_bf, edges, edge_mask,
        eW1T + (size_t)r * HID_ * KP1E, eb1 + (size_t)r * HID_,
        eW2T + (size_t)r * HID_ * KP2,  eb2 + (size_t)r * HID_, agg);
    node_mlp_kernel<<<128, 256, 0, stream>>>(
        h_f32, h_bf, agg, node_mask,
        nW1T + (size_t)r * HID_ * KP1N, nb1 + (size_t)r * HID_,
        nW2T + (size_t)r * HID_ * KP2,  nb2 + (size_t)r * HID_);
  }
  finalize_kernel<<<2048, 256, 0, stream>>>(h_f32, marker, (float*)d_out);
}
